// GC_FFM_39109972197410
// MI455X (gfx1250) — matrix operand order repaired and checked by reading, not compiled, not run on hardware
//
#include <hip/hip_runtime.h>

typedef __attribute__((ext_vector_type(16))) _Float16 v16h;
typedef __attribute__((ext_vector_type(8)))  _Float16 v8h;
typedef __attribute__((ext_vector_type(4)))  _Float16 v4h;
typedef __attribute__((ext_vector_type(8)))  float    v8f;
typedef __attribute__((ext_vector_type(4)))  float    v4f;

#define HW 4096

__device__ __forceinline__ v8f wmma_f16(v16h a, v16h b, v8f c) {
    // 8 args: (neg_a, A, neg_b, B, c_mod, C, reuse_a, reuse_b)
    return __builtin_amdgcn_wmma_f32_16x16x32_f16(false, a, false, b, (short)0, c, false, false);
}

// A fragment (16x32 f16): lane M = lane&15, K base = (lane>>4)*16.
// elements 0..15 = K 2*kb..2*kb+15 (contiguous).
__device__ __forceinline__ v16h load_frag_a(const _Float16* row, int kb) {
    v8h lo = *(const v8h*)(row + 2 * kb);
    v8h hi = *(const v8h*)(row + 2 * kb + 8);
    return __builtin_shufflevector(lo, hi, 0,1,2,3,4,5,6,7,8,9,10,11,12,13,14,15);
}

// ---------------------------------------------------------------------------
// Projection GEMM from fp32 features: out = W[O][Cin] * X[n][Cin][HW] + b.
// om==1: f16 output [n][O][hw] (row-major, for V / A-operand use)
// om==2: f16 output TRANSPOSED [n][hw][O] (channel-contiguous, b128 stores)
// ---------------------------------------------------------------------------
template<int BM>
__global__ void __launch_bounds__(256)
k_proj(const float* __restrict__ W, const float* __restrict__ X,
       const float* __restrict__ bias, _Float16* __restrict__ out,
       int O, int Cin, int om)
{
    const int j0 = blockIdx.x * 64;
    const int o0 = blockIdx.y * BM;
    const int n  = blockIdx.z;
    X += (size_t)n * Cin * HW;

    const int t = threadIdx.x;
    const int lane = t & 31, wave = t >> 5;
    const int nn = lane & 15;
    const int kb  = (lane >> 4) * 8;
    const int kb2 = (lane >> 4) * 16;

    __shared__ alignas(64) _Float16 A_lds[BM][64];   // [o][k]
    __shared__ alignas(64) _Float16 BT_lds[64][64];  // [p][k]

    constexpr int NT = (BM == 64) ? 2 : 1;
    const int msub  = (BM == 64) ? (wave & 3) : (wave & 1);
    const int jsub0 = (BM == 64) ? (wave >> 2) : (wave >> 1);

    v8f acc[NT];
    #pragma unroll
    for (int q = 0; q < NT; ++q) acc[q] = (v8f){};

    for (int c0 = 0; c0 < Cin; c0 += 64) {
        __syncthreads();
        constexpr int AREP = BM / 16;
        #pragma unroll
        for (int rep = 0; rep < AREP; ++rep) {       // A: v4f -> v4h
            int idx = t + rep * 256;
            int m = idx >> 4, k4 = (idx & 15) * 4;
            v4f v = *(const v4f*)(W + (size_t)(o0 + m) * Cin + c0 + k4);
            v4h h; h[0]=(_Float16)v[0]; h[1]=(_Float16)v[1];
                   h[2]=(_Float16)v[2]; h[3]=(_Float16)v[3];
            *(v4h*)(&A_lds[m][k4]) = h;
        }
        #pragma unroll
        for (int rep = 0; rep < 2; ++rep) {          // B: transpose+convert
            int g = t + rep * 256;                   // 0..511
            int kp = g >> 4, jj4 = (g & 15) * 4;     // k = 2*kp
            const float* xr = X + (size_t)(c0 + 2 * kp) * HW + j0 + jj4;
            v4f r0 = *(const v4f*)xr;
            v4f r1 = *(const v4f*)(xr + HW);
            #pragma unroll
            for (int q = 0; q < 4; ++q) {
                v4h h2; h2[0] = (_Float16)r0[q]; h2[1] = (_Float16)r1[q];
                *(__attribute__((ext_vector_type(2))) _Float16*)
                    (&BT_lds[jj4 + q][2 * kp]) =
                    (__attribute__((ext_vector_type(2))) _Float16){h2[0], h2[1]};
            }
        }
        if (c0 + 64 < Cin)
            __builtin_prefetch(X + (size_t)(c0 + 64 + (t >> 2)) * HW + j0, 0, 1);
        __syncthreads();

        const _Float16* arow = &A_lds[16 * msub + nn][0];
        v16h a0 = load_frag_a(arow, kb);
        v16h a1 = load_frag_a(arow + 32, kb);
        #pragma unroll
        for (int q = 0; q < NT; ++q) {
            const _Float16* brow = &BT_lds[16 * (jsub0 + 2 * q) + nn][0];
            v16h b0 = *(const v16h*)(brow + kb2);
            v16h b1 = *(const v16h*)(brow + 32 + kb2);
            acc[q] = wmma_f16(a0, b0, acc[q]);
            acc[q] = wmma_f16(a1, b1, acc[q]);
        }
    }

    const int obase = o0 + 16 * msub + 8 * (lane >> 4);
    #pragma unroll
    for (int q = 0; q < NT; ++q) {
        int j = j0 + 16 * (jsub0 + 2 * q) + nn;
        if (om == 2) {                 // transposed f16: one b128 store
            v8h h;
            #pragma unroll
            for (int e = 0; e < 8; ++e)
                h[e] = (_Float16)(acc[q][e] + bias[obase + e]);
            *(v8h*)(out + ((size_t)n * HW + j) * O + obase) = h;
        } else {                       // row-major f16
            #pragma unroll
            for (int e = 0; e < 8; ++e)
                out[(size_t)n * O * HW + (size_t)(obase + e) * HW + j] =
                    (_Float16)(acc[q][e] + bias[obase + e]);
        }
    }
}

// ---------------------------------------------------------------------------
// GEMM over pre-transposed f16 activations: Xt[n][hw][Cin].
// B staging = straight b128 copies (no transpose, no convert).
// om==0: f32 row-major output (+ optional addsrc); om==2: f16 transposed out.
// act: 0=none 1=relu 2=sigmoid.
// ---------------------------------------------------------------------------
__global__ void __launch_bounds__(256)
k_gemm16(const float* __restrict__ W, const _Float16* __restrict__ Xt,
         const float* __restrict__ bias, void* __restrict__ out,
         const float* __restrict__ addsrc, int O, int Cin, int act, int om)
{
    const int j0 = blockIdx.x * 64;
    const int o0 = blockIdx.y * 64;
    const int n  = blockIdx.z;
    Xt += (size_t)n * HW * Cin;

    const int t = threadIdx.x;
    const int lane = t & 31, wave = t >> 5;
    const int nn = lane & 15;
    const int kb  = (lane >> 4) * 8;
    const int kb2 = (lane >> 4) * 16;
    const int msub = wave & 3, jsub0 = wave >> 2;

    __shared__ alignas(64) _Float16 A_lds[64][64];
    __shared__ alignas(64) _Float16 BT_lds[64][64];

    v8f acc0 = {}, acc1 = {};

    for (int c0 = 0; c0 < Cin; c0 += 64) {
        __syncthreads();
        #pragma unroll
        for (int rep = 0; rep < 4; ++rep) {          // A: v4f -> v4h
            int idx = t + rep * 256;
            int m = idx >> 4, k4 = (idx & 15) * 4;
            v4f v = *(const v4f*)(W + (size_t)(o0 + m) * Cin + c0 + k4);
            v4h h; h[0]=(_Float16)v[0]; h[1]=(_Float16)v[1];
                   h[2]=(_Float16)v[2]; h[3]=(_Float16)v[3];
            *(v4h*)(&A_lds[m][k4]) = h;
        }
        #pragma unroll
        for (int rep = 0; rep < 2; ++rep) {          // B: straight b128 copy
            int idx = t + rep * 256;                 // 0..511
            int jj = idx >> 3, seg = (idx & 7) * 8;
            *(v8h*)(&BT_lds[jj][seg]) =
                *(const v8h*)(Xt + (size_t)(j0 + jj) * Cin + c0 + seg);
        }
        if (c0 + 64 < Cin)
            __builtin_prefetch(Xt + (size_t)(j0 + (t >> 2)) * Cin + c0 + 64, 0, 1);
        __syncthreads();

        const _Float16* arow = &A_lds[16 * msub + nn][0];
        v16h a0 = load_frag_a(arow, kb);
        v16h a1 = load_frag_a(arow + 32, kb);
        {
            const _Float16* brow = &BT_lds[16 * jsub0 + nn][0];
            acc0 = wmma_f16(a0, *(const v16h*)(brow + kb2), acc0);
            acc0 = wmma_f16(a1, *(const v16h*)(brow + 32 + kb2), acc0);
        }
        {
            const _Float16* brow = &BT_lds[16 * (jsub0 + 2) + nn][0];
            acc1 = wmma_f16(a0, *(const v16h*)(brow + kb2), acc1);
            acc1 = wmma_f16(a1, *(const v16h*)(brow + 32 + kb2), acc1);
        }
    }

    const int obase = o0 + 16 * msub + 8 * (lane >> 4);
    v8f accs[2] = {acc0, acc1};
    #pragma unroll
    for (int q = 0; q < 2; ++q) {
        int j = j0 + 16 * (jsub0 + 2 * q) + nn;
        if (om == 2) {
            v8h h;
            #pragma unroll
            for (int e = 0; e < 8; ++e) {
                float v = accs[q][e] + bias[obase + e];
                if (act == 1) v = fmaxf(v, 0.f);
                else if (act == 2) v = 1.f / (1.f + __expf(-v));
                h[e] = (_Float16)v;
            }
            *(v8h*)((_Float16*)out + ((size_t)n * HW + j) * O + obase) = h;
        } else {
            #pragma unroll
            for (int e = 0; e < 8; ++e) {
                float v = accs[q][e] + bias[obase + e];
                if (act == 1) v = fmaxf(v, 0.f);
                else if (act == 2) v = 1.f / (1.f + __expf(-v));
                size_t oidx = (size_t)n * O * HW + (size_t)(obase + e) * HW + j;
                if (addsrc) v += addsrc[oidx];
                ((float*)out)[oidx] = v;
            }
        }
    }
}

// ---------------------------------------------------------------------------
// Attention pass 1: softmax stats of S = Q K^T rows. Qt,Kt: f16 [n][hw][32]
// (channel-contiguous) -> A/B fragments are direct aligned global loads.
// NO LDS, NO barriers. Online max/sum + shfl_xor merge (wave32).
// ---------------------------------------------------------------------------
__global__ void __launch_bounds__(256)
k_attn_pass1(const _Float16* __restrict__ Qt, const _Float16* __restrict__ Kt,
             float* __restrict__ Mrow, float* __restrict__ Zinv)
{
    const int n = blockIdx.z;
    Qt += (size_t)n * HW * 32;  Kt += (size_t)n * HW * 32;
    Mrow += (size_t)n * HW;     Zinv += (size_t)n * HW;
    const int iblk = blockIdx.x * 128;

    const int t = threadIdx.x, lane = t & 31, wave = t >> 5;
    const int nn = lane & 15, kb = (lane >> 4) * 8, kb2 = (lane >> 4) * 16;

    v16h a = load_frag_a(Qt + (size_t)(iblk + wave * 16 + nn) * 32, kb);

    float m[8], z[8];
    #pragma unroll
    for (int e = 0; e < 8; ++e) { m[e] = -1e30f; z[e] = 0.f; }

    for (int j0 = 0; j0 < HW; j0 += 64) {
        __builtin_prefetch(Kt + (size_t)(j0 + 64 + lane) * 32, 0, 1);
        #pragma unroll
        for (int s = 0; s < 4; ++s) {
            v16h b = *(const v16h*)(Kt + (size_t)(j0 + 16 * s + nn) * 32 + kb2);
            v8f c = {};
            c = wmma_f16(a, b, c);
            #pragma unroll
            for (int e = 0; e < 8; ++e) {            // online softmax update
                float sv = c[e];
                float nm = fmaxf(m[e], sv);
                z[e] = z[e] * __expf(m[e] - nm) + __expf(sv - nm);
                m[e] = nm;
            }
        }
    }
    #pragma unroll
    for (int e = 0; e < 8; ++e) {                    // merge 16 column-stripes
        float me = m[e], ze = z[e];
        for (int off = 1; off < 16; off <<= 1) {
            float mo = __shfl_xor(me, off, 32);
            float zo = __shfl_xor(ze, off, 32);
            float nm = fmaxf(me, mo);
            ze = ze * __expf(me - nm) + zo * __expf(mo - nm);
            me = nm;
        }
        if (nn == 0) {
            int i = iblk + wave * 16 + 8 * (lane >> 4) + e;
            Mrow[i] = me;
            Zinv[i] = 1.f / ze;
        }
    }
}

// ---------------------------------------------------------------------------
// Attention pass 2: outT[j][c] = resid[c][j] + sum_i V[c][i]*softmax(S)[i][j].
// Qt,Kt direct-global fragments; only the probability tile lives in LDS.
// Output written TRANSPOSED f16 into BRBDt[n][hw][512] (+branch offset).
// ---------------------------------------------------------------------------
__global__ void __launch_bounds__(256)
k_attn_pass2(const _Float16* __restrict__ Qt, const _Float16* __restrict__ Kt,
             const _Float16* __restrict__ V,
             const float* __restrict__ Mrow, const float* __restrict__ Zinv,
             const float* __restrict__ resid, _Float16* __restrict__ outT,
             int branchOff)
{
    const int C = 256;
    const int n = blockIdx.z;
    Qt += (size_t)n * HW * 32;  Kt += (size_t)n * HW * 32;  V += (size_t)n * C * HW;
    Mrow += (size_t)n * HW;     Zinv += (size_t)n * HW;
    resid += (size_t)n * C * HW;
    outT  += (size_t)n * HW * 512 + branchOff;

    const int j0 = blockIdx.x * 64;
    const int c0 = blockIdx.y * 64;
    const int t = threadIdx.x, lane = t & 31, wave = t >> 5;
    const int nn = lane & 15, kb = (lane >> 4) * 8, kb2 = (lane >> 4) * 16;
    const int csub = wave & 3, jhalf = wave >> 2;    // out-phase tiles
    const int msub_s = wave >> 2, jsub_s = wave & 3; // S-phase tile

    __shared__ alignas(64) _Float16 PT[64][32];      // [j][i] probabilities

    v16h bk = *(const v16h*)(Kt + (size_t)(j0 + 16 * jsub_s + nn) * 32 + kb2);

    v8f acc0 = {}, acc1 = {};

    for (int i0 = 0; i0 < HW; i0 += 32) {
        __syncthreads();                             // PT readers done
        {   // S-phase: score tile -> probabilities in PT
            v16h aq = load_frag_a(Qt + (size_t)(i0 + 16 * msub_s + nn) * 32, kb);
            v8f s = {};
            s = wmma_f16(aq, bk, s);
            int ibase = i0 + 16 * msub_s + 8 * (lane >> 4);
            v8h p;
            #pragma unroll
            for (int e = 0; e < 8; ++e) {
                float mi = Mrow[ibase + e];
                float zi = Zinv[ibase + e];
                p[e] = (_Float16)(__expf(s[e] - mi) * zi);
            }
            *(v8h*)(&PT[16 * jsub_s + nn][16 * msub_s + 8 * (lane >> 4)]) = p;
        }
        __syncthreads();
        {   // out-phase: acc += V_tile(16c x 32i) x P(32i x 16j)
            const _Float16* vrow = V + (size_t)(c0 + 16 * csub + nn) * HW + i0;
            v16h av  = load_frag_a(vrow, kb);
            v16h bp0 = *(const v16h*)(&PT[16 * jhalf + nn][kb2]);
            v16h bp1 = *(const v16h*)(&PT[16 * (jhalf + 2) + nn][kb2]);
            acc0 = wmma_f16(av, bp0, acc0);
            acc1 = wmma_f16(av, bp1, acc1);
        }
    }

    const int cbase = c0 + 16 * csub + 8 * (lane >> 4);
    const int jA = j0 + 16 * jhalf + nn;
    const int jB = j0 + 16 * (jhalf + 2) + nn;
    v8h hA, hB;
    #pragma unroll
    for (int e = 0; e < 8; ++e) {
        int c = cbase + e;
        hA[e] = (_Float16)(acc0[e] + resid[(size_t)c * HW + jA]);
        hB[e] = (_Float16)(acc1[e] + resid[(size_t)c * HW + jB]);
    }
    *(v8h*)(outT + (size_t)jA * 512 + cbase) = hA;
    *(v8h*)(outT + (size_t)jB * 512 + cbase) = hB;
}

// ---------------------------------------------------------------------------
// gate fuse over channel-contiguous f16 tensors (fully coalesced in c):
// Ft[p][c] = br*g + bd*(1-g);  Ft[p][256+c] = br*bd
// ---------------------------------------------------------------------------
__global__ void __launch_bounds__(256)
k_gate_fuse(const _Float16* __restrict__ Gt, const _Float16* __restrict__ BRBDt,
            _Float16* __restrict__ Ft)
{
    size_t idx = (size_t)blockIdx.x * blockDim.x + threadIdx.x;
    if (idx >= (size_t)2 * HW * 256) return;
    size_t n = idx / ((size_t)HW * 256);
    size_t rem = idx % ((size_t)HW * 256);
    size_t p = rem >> 8, c = rem & 255;
    size_t gi = (n * HW + p) * 256 + c;
    size_t bi = (n * HW + p) * 512 + c;
    float g  = (float)Gt[gi];
    float br = (float)BRBDt[bi];
    float bd = (float)BRBDt[bi + 256];
    Ft[bi]       = (_Float16)(br * g + bd * (1.f - g));
    Ft[bi + 256] = (_Float16)(br * bd);
}

// ---------------------------------------------------------------------------
// 3x3 conv (512->512, SAME) + relu, implicit GEMM over f16 channel-contiguous
// input Y1t[n][hw][512]; input staging = straight b128 copies with halo;
// output transposed f16 Y2t[n][hw][512].
// ---------------------------------------------------------------------------
__global__ void __launch_bounds__(256)
k_conv3x3(const _Float16* __restrict__ Xt, const float* __restrict__ Wt,
          const float* __restrict__ bias, _Float16* __restrict__ Yt)
{
    const int Cio = 512;
    const int y  = blockIdx.x;            // image row 0..63
    const int o0 = blockIdx.y * 64;
    const int n  = blockIdx.z;
    Xt += (size_t)n * HW * Cio;
    Yt += (size_t)n * HW * Cio;

    const int t = threadIdx.x, lane = t & 31, wave = t >> 5;
    const int nn = lane & 15, kb = (lane >> 4) * 8, kb2 = (lane >> 4) * 16;
    const int osub = wave & 3, jhalf = wave >> 2;

    __shared__ alignas(64) _Float16 InT[66][32];     // [x+1][i], halo cols 0,65
    __shared__ alignas(64) _Float16 A3[3][64][32];   // [kx][o][i]

    v8f acc0 = {}, acc1 = {};

    for (int i0 = 0; i0 < Cio; i0 += 32) {
        for (int ky = 0; ky < 3; ++ky) {
            __syncthreads();
            const int r = y + ky - 1;
            {   // input row: one b128 copy per thread
                int x = t >> 2, seg = (t & 3) * 8;
                v8h v = {};
                if (r >= 0 && r < 64)
                    v = *(const v8h*)(Xt + (size_t)(r * 64 + x) * Cio + i0 + seg);
                *(v8h*)(&InT[x + 1][seg]) = v;
            }
            if (t < 32) { InT[0][t] = (_Float16)0.f; InT[65][t] = (_Float16)0.f; }
            #pragma unroll
            for (int rep = 0; rep < 12; ++rep) {     // weights [kx][o][i] pairs
                int g = t + rep * 256;               // 0..3071
                int kx = g / 1024;
                int rem = g & 1023;
                int o = rem >> 4, ip = rem & 15;     // i = 2*ip
                const float* wsrc =
                    Wt + ((size_t)(o0 + o) * Cio + i0 + 2 * ip) * 9 + ky * 3 + kx;
                *(__attribute__((ext_vector_type(2))) _Float16*)(&A3[kx][o][2 * ip]) =
                    (__attribute__((ext_vector_type(2))) _Float16){
                        (_Float16)wsrc[0], (_Float16)wsrc[9]};
            }
            __syncthreads();
            #pragma unroll
            for (int kx = 0; kx < 3; ++kx) {
                v16h a  = load_frag_a(&A3[kx][16 * osub + nn][0], kb);
                v16h b0 = *(const v16h*)(&InT[16 * jhalf + nn + kx][kb2]);
                v16h b1 = *(const v16h*)(&InT[16 * (jhalf + 2) + nn + kx][kb2]);
                acc0 = wmma_f16(a, b0, acc0);
                acc1 = wmma_f16(a, b1, acc1);
            }
        }
    }

    const int obase = o0 + 16 * osub + 8 * (lane >> 4);
    const int posA = y * 64 + 16 * jhalf + nn;
    const int posB = y * 64 + 16 * (jhalf + 2) + nn;
    v8h hA, hB;
    #pragma unroll
    for (int e = 0; e < 8; ++e) {
        float b = bias[obase + e];
        hA[e] = (_Float16)fmaxf(acc0[e] + b, 0.f);
        hB[e] = (_Float16)fmaxf(acc1[e] + b, 0.f);
    }
    *(v8h*)(Yt + (size_t)posA * Cio + obase) = hA;
    *(v8h*)(Yt + (size_t)posB * Cio + obase) = hB;
}

// ---------------------------------------------------------------------------
extern "C" void kernel_launch(void* const* d_in, const int* in_sizes, int n_in,
                              void* d_out, int out_size, void* d_ws, size_t ws_size,
                              hipStream_t stream)
{
    (void)in_sizes; (void)n_in; (void)out_size; (void)ws_size;
    const float* rgb  = (const float*)d_in[0];
    const float* chm  = (const float*)d_in[1];
    const float* rq_w = (const float*)d_in[2];  const float* rq_b = (const float*)d_in[3];
    const float* rk_w = (const float*)d_in[4];  const float* rk_b = (const float*)d_in[5];
    const float* rv_w = (const float*)d_in[6];  const float* rv_b = (const float*)d_in[7];
    const float* dq_w = (const float*)d_in[8];  const float* dq_b = (const float*)d_in[9];
    const float* dk_w = (const float*)d_in[10]; const float* dk_b = (const float*)d_in[11];
    const float* dv_w = (const float*)d_in[12]; const float* dv_b = (const float*)d_in[13];
    const float* g_w  = (const float*)d_in[14]; const float* g_b  = (const float*)d_in[15];
    const float* f1_w = (const float*)d_in[16]; const float* f1_b = (const float*)d_in[17];
    const float* f2_w = (const float*)d_in[18]; const float* f2_b = (const float*)d_in[19];
    const float* f3_w = (const float*)d_in[20]; const float* f3_b = (const float*)d_in[21];
    const float* sk_w = (const float*)d_in[22]; const float* sk_b = (const float*)d_in[23];
    float* out = (float*)d_out;

    size_t off = 0;
    auto carve = [&](size_t bytes) {
        void* p = (char*)d_ws + off;
        off += (bytes + 255) & ~(size_t)255;
        return p;
    };
    _Float16* Qtr = (_Float16*)carve((size_t)2 * HW * 32 * 2);   // [n][hw][32]
    _Float16* Ktr = (_Float16*)carve((size_t)2 * HW * 32 * 2);
    _Float16* Qtd = (_Float16*)carve((size_t)2 * HW * 32 * 2);
    _Float16* Ktd = (_Float16*)carve((size_t)2 * HW * 32 * 2);
    _Float16* Vr  = (_Float16*)carve((size_t)2 * 256 * HW * 2);  // [n][256][hw]
    _Float16* Vd  = (_Float16*)carve((size_t)2 * 256 * HW * 2);
    float* Mr = (float*)carve((size_t)2 * HW * 4);
    float* Zr = (float*)carve((size_t)2 * HW * 4);
    float* Md = (float*)carve((size_t)2 * HW * 4);
    float* Zd = (float*)carve((size_t)2 * HW * 4);
    _Float16* BRBDt = (_Float16*)carve((size_t)2 * HW * 512 * 2); // [n][hw][512]
    _Float16* Gt    = (_Float16*)carve((size_t)2 * HW * 256 * 2); // [n][hw][256]
    _Float16* Ft    = (_Float16*)carve((size_t)2 * HW * 512 * 2);
    _Float16* Y1t   = (_Float16*)carve((size_t)2 * HW * 512 * 2);
    _Float16* Y2t   = (_Float16*)carve((size_t)2 * HW * 512 * 2);
    float* SK = (float*)carve((size_t)2 * 256 * HW * 4);          // [n][256][hw]

    const dim3 blk(256);

    // projections: q/k stored transposed [hw][32]; v row-major [256][hw]
    k_proj<32><<<dim3(64, 1, 2), blk, 0, stream>>>(rq_w, rgb, rq_b, Qtr, 32, 256, 2);
    k_proj<32><<<dim3(64, 1, 2), blk, 0, stream>>>(rk_w, rgb, rk_b, Ktr, 32, 256, 2);
    k_proj<64><<<dim3(64, 4, 2), blk, 0, stream>>>(rv_w, rgb, rv_b, Vr, 256, 256, 1);
    k_proj<32><<<dim3(64, 1, 2), blk, 0, stream>>>(dq_w, chm, dq_b, Qtd, 32, 256, 2);
    k_proj<32><<<dim3(64, 1, 2), blk, 0, stream>>>(dk_w, chm, dk_b, Ktd, 32, 256, 2);
    k_proj<64><<<dim3(64, 4, 2), blk, 0, stream>>>(dv_w, chm, dv_b, Vd, 256, 256, 1);

    // softmax stats (rgb branch attends with depth q/k, depth with rgb q/k)
    k_attn_pass1<<<dim3(32, 1, 2), blk, 0, stream>>>(Qtd, Ktd, Md, Zd);
    k_attn_pass1<<<dim3(32, 1, 2), blk, 0, stream>>>(Qtr, Ktr, Mr, Zr);

    // attended branches + residual -> concatenated BRBDt (f16, [hw][512])
    k_attn_pass2<<<dim3(64, 4, 2), blk, 0, stream>>>(Qtd, Ktd, Vr, Md, Zd, rgb, BRBDt, 0);
    k_attn_pass2<<<dim3(64, 4, 2), blk, 0, stream>>>(Qtr, Ktr, Vd, Mr, Zr, chm, BRBDt, 256);

    // gate (sigmoid fused, transposed f16 out), elementwise fuse -> Ft
    k_gemm16<<<dim3(64, 4, 2), blk, 0, stream>>>(g_w, BRBDt, g_b, Gt, nullptr, 256, 512, 2, 2);
    k_gate_fuse<<<dim3(8192), blk, 0, stream>>>(Gt, BRBDt, Ft);

    // fusion trunk
    k_gemm16<<<dim3(64, 8, 2), blk, 0, stream>>>(f1_w, Ft, f1_b, Y1t, nullptr, 512, 512, 1, 2);
    k_conv3x3<<<dim3(64, 8, 2), blk, 0, stream>>>(Y1t, f2_w, f2_b, Y2t);
    k_gemm16<<<dim3(64, 4, 2), blk, 0, stream>>>(sk_w, Ft, sk_b, SK, nullptr, 256, 512, 0, 0);
    k_gemm16<<<dim3(64, 4, 2), blk, 0, stream>>>(f3_w, Y2t, f3_b, out, SK, 256, 512, 0, 0);
}
